// AttentionBlock_69569880261190
// MI455X (gfx1250) — matrix operand order repaired and checked by reading, not compiled, not run on hardware
//
#include <hip/hip_runtime.h>
#include <math.h>

// ---------------- types ----------------
typedef __bf16 bf16;
typedef bf16  v16bf __attribute__((ext_vector_type(16)));
typedef bf16  v8bf  __attribute__((ext_vector_type(8)));
typedef float v8f   __attribute__((ext_vector_type(8)));
typedef int   v4i_vs __attribute__((vector_size(4 * sizeof(int))));  // b128 payload

#define T_TOK   1024
#define HIDDEN  2880
#define HD      64
#define NH      64
#define NKV     8
#define SW      128
#define QKVD    5120   // HD*(NH+2*NKV)
#define QOFF    0
#define KOFF    4096   // NH*HD
#define VOFF    4608   // NH*HD + NKV*HD
#define ATTND   4096   // NH*HD

#if defined(__has_builtin)
#if __has_builtin(__builtin_amdgcn_global_load_async_to_lds_b128)
#define HAVE_ASYNC_G2L 1
#endif
#if __has_builtin(__builtin_amdgcn_s_wait_asynccnt)
#define HAVE_WAIT_ASYNC_BUILTIN 1
#endif
#endif

__device__ __forceinline__ bf16 f2bf(float f) {
    unsigned u = __builtin_bit_cast(unsigned, f);
    u += 0x7FFFu + ((u >> 16) & 1u);           // round-to-nearest-even
    unsigned short h = (unsigned short)(u >> 16);
    return __builtin_bit_cast(bf16, h);
}

// two floats -> packed bf16 pair (low = a, high = b)
__device__ __forceinline__ unsigned pk2bf(float a, float b) {
    unsigned short lo = __builtin_bit_cast(unsigned short, f2bf(a));
    unsigned short hi = __builtin_bit_cast(unsigned short, f2bf(b));
    return (unsigned)lo | ((unsigned)hi << 16);
}

// async DMA: 16B global -> LDS, tracked by ASYNCcnt (gfx1250 path)
__device__ __forceinline__ void async_g2l_b128(const void* g, void* l) {
#if defined(HAVE_ASYNC_G2L)
    __builtin_amdgcn_global_load_async_to_lds_b128(
        (__attribute__((address_space(1))) v4i_vs*)(void*)g,
        (__attribute__((address_space(3))) v4i_vs*)l, 0, 0);
#else
    *(v8bf*)l = *(const v8bf*)g;
#endif
}

__device__ __forceinline__ void wait_async0() {
#if defined(HAVE_ASYNC_G2L)
#if defined(HAVE_WAIT_ASYNC_BUILTIN)
    __builtin_amdgcn_s_wait_asynccnt(0);
#else
    asm volatile("s_wait_asynccnt 0" ::: "memory");
#endif
#endif
}

__device__ __forceinline__ v8f wmma_bf16(v16bf a, v16bf b, v8f c) {
    // v_wmma_f32_16x16x32_bf16: (neg_a, A, neg_b, B, c_mod, C, reuse_a, reuse_b)
    return __builtin_amdgcn_wmma_f32_16x16x32_bf16(false, a, false, b, (short)0, c,
                                                   false, false);
}

// A-matrix 16x32 bf16 fragment from row-major LDS tile [row][K-contig].
// ISA layout: lanes 0-15 / 16-31 hold M=0..15; VGPR0-3: K = half*8 .. +7,
// VGPR4-7: K = 16 + half*8 .. +7  => two contiguous 8-elt (16B) chunks.
__device__ __forceinline__ v16bf frag_a(const bf16* base, int stride, int row0, int kbase) {
    const int lane = threadIdx.x & 31;
    const bf16* p = base + (row0 + (lane & 15)) * stride + kbase + ((lane >> 4) << 3);
    v8bf lo = *(const v8bf*)(p);
    v8bf hi = *(const v8bf*)(p + 16);
    v16bf r;
#pragma unroll
    for (int i = 0; i < 8; ++i) { r[i] = lo[i]; r[i + 8] = hi[i]; }
    return r;
}

// B-matrix 32x16 bf16 fragment from row-major LDS tile [N][K-contig].
// ISA layout: lane holds N=lane%16, K = half*8 .. +7 and 16 + half*8 .. +7 (same as A).
__device__ __forceinline__ v16bf frag_b(const bf16* base, int stride, int row0, int kbase) {
    const int lane = threadIdx.x & 31;
    const bf16* p = base + (row0 + (lane & 15)) * stride + kbase + ((lane >> 4) << 3);
    v8bf lo = *(const v8bf*)(p);
    v8bf hi = *(const v8bf*)(p + 16);
    v16bf r;
#pragma unroll
    for (int i = 0; i < 8; ++i) { r[i] = lo[i]; r[i + 8] = hi[i]; }
    return r;
}

// ---------------- one-shot weight conversion: f32 -> bf16 (runs once per launch) ----------------
__global__ __launch_bounds__(256) void cvt_w_kernel(const float* __restrict__ src,
                                                    bf16* __restrict__ dst, long n4) {
    long i = (long)blockIdx.x * 256 + threadIdx.x;
    const long stride = (long)gridDim.x * 256;
    for (; i < n4; i += stride) {
        float4 f = *(const float4*)(src + i * 4);
        uint2 pk;
        pk.x = pk2bf(f.x, f.y);
        pk.y = pk2bf(f.z, f.w);
        *(uint2*)(dst + i * 4) = pk;
    }
}

// ---------------- RMSNorm: x (1024x2880) -> t bf16 ----------------
__global__ __launch_bounds__(256) void rmsnorm_kernel(const float* __restrict__ x,
                                                      const float* __restrict__ g,
                                                      bf16* __restrict__ tb) {
    __shared__ float red[8];
    const int row = blockIdx.x;
    const int tid = threadIdx.x;
    const float* xr = x + (long)row * HIDDEN;
    float ss = 0.f;
    for (int i = tid; i < HIDDEN; i += 256) { float v = xr[i]; ss += v * v; }
#pragma unroll
    for (int off = 16; off > 0; off >>= 1) ss += __shfl_xor(ss, off, 32);
    if ((tid & 31) == 0) red[tid >> 5] = ss;
    __syncthreads();
    float tot = 0.f;
#pragma unroll
    for (int i = 0; i < 8; ++i) tot += red[i];
    const float rs = rsqrtf(tot / (float)HIDDEN + 1e-5f);
    for (int i = tid * 2; i < HIDDEN; i += 512) {
        unsigned pk = pk2bf(xr[i] * rs * g[i], xr[i + 1] * rs * g[i + 1]);
        *(unsigned*)(tb + (long)row * HIDDEN + i) = pk;
    }
}

// ---------------- RoPE tables (YaRN-style) ----------------
__global__ __launch_bounds__(256) void rope_tables_kernel(float* __restrict__ cosT,
                                                          float* __restrict__ sinT) {
    const int idx = blockIdx.x * 256 + threadIdx.x;
    if (idx >= T_TOK * 32) return;
    const int t = idx >> 5, i = idx & 31;
    const float base = 150000.f;
    const float two_pi = 6.283185307179586f;
    const float freq = powf(base, (float)(2 * i) / 64.f);
    const float interp = 1.f / (32.f * freq);
    const float extrap = 1.f / freq;
    const float lb = logf(base);
    const float low  = 32.f * logf(4096.f / (32.f * two_pi)) / lb;
    const float high = 32.f * logf(4096.f / (1.f * two_pi)) / lb;
    float ramp = ((float)i - low) / (high - low);
    ramp = fminf(fmaxf(ramp, 0.f), 1.f);
    const float m = 1.f - ramp;
    const float invf = interp * (1.f - m) + extrap * m;
    const float conc = 0.1f * logf(32.f) + 1.f;
    const float ang = (float)t * invf;
    cosT[idx] = cosf(ang) * conc;
    sinT[idx] = sinf(ang) * conc;
}

// ---------------- GEMM: C(MxN,f32) = A(MxK,bf16) * Bw(NxK,bf16)^T + bias ----------------
// Block tile 128x128, 8 waves (4x2), wave tile 32x64 = 2x4 accum tiles, K step 32.
// Double-buffered; BOTH operands staged by async global->LDS DMA (pure DMA + WMMA loop).
#define LSTR 40   // LDS K-stride (bf16 elems): 80B, 16B aligned
__global__ __launch_bounds__(256) void gemm_bf16_kernel(const bf16* __restrict__ A,
                                                        const bf16* __restrict__ Bw,
                                                        const float* __restrict__ bias,
                                                        float* __restrict__ C,
                                                        int M, int N, int K) {
    __shared__ bf16 As[2][128 * LSTR];
    __shared__ bf16 Bs[2][128 * LSTR];
    const int tid = threadIdx.x;
    const int lane = tid & 31, w = tid >> 5, half = lane >> 4;
    const int bm = blockIdx.y * 128, bn = blockIdx.x * 128;
    const int wm = (w & 3) * 32, wn = (w >> 2) * 64;
    const int nk = K >> 5;
    // staging coordinates: each thread owns two 16B chunks per 128x32 tile
    const int r0 = tid >> 2, r1 = (tid + 256) >> 2;
    const int cc = (tid & 3) << 3;

    v8f acc[2][4];
#pragma unroll
    for (int mt = 0; mt < 2; ++mt)
#pragma unroll
        for (int nt = 0; nt < 4; ++nt)
#pragma unroll
            for (int r = 0; r < 8; ++r) acc[mt][nt][r] = 0.f;

    // ---- prologue: stage k-step 0 into buffer 0 (all async DMA) ----
    async_g2l_b128(A + (long)(bm + r0) * K + cc, &As[0][r0 * LSTR + cc]);
    async_g2l_b128(A + (long)(bm + r1) * K + cc, &As[0][r1 * LSTR + cc]);
    if (bn + r0 < N) async_g2l_b128(Bw + (long)(bn + r0) * K + cc, &Bs[0][r0 * LSTR + cc]);
    if (bn + r1 < N) async_g2l_b128(Bw + (long)(bn + r1) * K + cc, &Bs[0][r1 * LSTR + cc]);
    wait_async0();
    __syncthreads();

    // ---- pipelined main loop: async-stage next tile while WMMAs run on current ----
    for (int kt = 0; kt < nk; ++kt) {
        const int cur = kt & 1, nxt = cur ^ 1;
        const bool pend = (kt + 1 < nk);
        const int k1 = (kt + 1) << 5;
        if (pend) {
            async_g2l_b128(A + (long)(bm + r0) * K + k1 + cc, &As[nxt][r0 * LSTR + cc]);
            async_g2l_b128(A + (long)(bm + r1) * K + k1 + cc, &As[nxt][r1 * LSTR + cc]);
            if (bn + r0 < N) {
                const bf16* gb = Bw + (long)(bn + r0) * K + k1 + cc;
                async_g2l_b128(gb, &Bs[nxt][r0 * LSTR + cc]);
                __builtin_prefetch(gb + 32, 0, 0);   // K-step kt+2 (speculative)
            }
            if (bn + r1 < N) {
                async_g2l_b128(Bw + (long)(bn + r1) * K + k1 + cc,
                               &Bs[nxt][r1 * LSTR + cc]);
            }
        }
        // compute on current buffer
        v16bf a0 = frag_a(&As[cur][0], LSTR, wm + 0,  0);
        v16bf a1 = frag_a(&As[cur][0], LSTR, wm + 16, 0);
#pragma unroll
        for (int nt = 0; nt < 4; ++nt) {
            v16bf b = frag_b(&Bs[cur][0], LSTR, wn + nt * 16, 0);
            acc[0][nt] = wmma_bf16(a0, b, acc[0][nt]);
            acc[1][nt] = wmma_bf16(a1, b, acc[1][nt]);
        }
        wait_async0();
        __syncthreads();
    }

#pragma unroll
    for (int mt = 0; mt < 2; ++mt)
#pragma unroll
        for (int nt = 0; nt < 4; ++nt)
#pragma unroll
            for (int r = 0; r < 8; ++r) {
                int row = bm + wm + mt * 16 + r + half * 8;
                int col = bn + wn + nt * 16 + (lane & 15);
                if (col < N) C[(long)row * N + col] = acc[mt][nt][r] + bias[col];
            }
}

// ---------------- Fused RoPE + sliding-window attention w/ sink ----------------
// Grid: (8 query blocks of 128, 64 heads). 8 waves; wave owns 16 query rows.
#define QSTR 72   // 64 + 8 pad (144B, 16B aligned)
__global__ __launch_bounds__(256) void attn_kernel(const float* __restrict__ qkv,
                                                   const float* __restrict__ cosT,
                                                   const float* __restrict__ sinT,
                                                   const float* __restrict__ sinks,
                                                   bf16* __restrict__ attn) {
    __shared__ bf16 Qs[128 * QSTR];  // [query][d]
    __shared__ bf16 Ks[64 * QSTR];   // [key][d]
    __shared__ bf16 Vt[64 * QSTR];   // [d][key] (transposed)
    __shared__ bf16 Ps[128 * QSTR];  // [query][key]
    const int qb = blockIdx.x, h = blockIdx.y, kvh = h >> 3;
    const int tid = threadIdx.x, lane = tid & 31, w = tid >> 5, half = lane >> 4;

    // ---- load Q tile (128x64) with RoPE, scale folded (1/sqrt(64)) ----
#pragma unroll
    for (int it = 0; it < 8; ++it) {
        int p = tid + it * 256;                 // 128 rows * 16 pair-pairs
        int row = p >> 4, ii = (p & 15) << 1;
        int t = qb * 128 + row;
        const float* qp = qkv + (long)t * QKVD + QOFF + h * HD;
        float2 q0 = *(const float2*)(qp + ii);
        float2 q1 = *(const float2*)(qp + ii + 32);
        float2 c  = *(const float2*)(cosT + t * 32 + ii);
        float2 s  = *(const float2*)(sinT + t * 32 + ii);
        *(unsigned*)(Qs + row * QSTR + ii) =
            pk2bf((q0.x * c.x - q1.x * s.x) * 0.125f, (q0.y * c.y - q1.y * s.y) * 0.125f);
        *(unsigned*)(Qs + row * QSTR + ii + 32) =
            pk2bf((q1.x * c.x + q0.x * s.x) * 0.125f, (q1.y * c.y + q0.y * s.y) * 0.125f);
    }

    v8f O[4];
#pragma unroll
    for (int nt = 0; nt < 4; ++nt)
#pragma unroll
        for (int r = 0; r < 8; ++r) O[nt][r] = 0.f;
    float m[8], l[8];
    const float sink = sinks[h];
#pragma unroll
    for (int r = 0; r < 8; ++r) { m[r] = sink; l[r] = 1.f; }

    const int kt0 = (2 * qb - 2 > 0) ? (2 * qb - 2) : 0;
    for (int kt = kt0; kt <= 2 * qb + 1; ++kt) {
        __syncthreads();   // prior iteration done reading Ks/Vt
        // ---- load K tile (64x64) with RoPE ----
#pragma unroll
        for (int it = 0; it < 4; ++it) {
            int p = tid + it * 256;             // 64 rows * 16 pair-pairs
            int row = p >> 4, ii = (p & 15) << 1;
            int t = kt * 64 + row;
            const float* kp = qkv + (long)t * QKVD + KOFF + kvh * HD;
            float2 k0 = *(const float2*)(kp + ii);
            float2 k1 = *(const float2*)(kp + ii + 32);
            float2 c  = *(const float2*)(cosT + t * 32 + ii);
            float2 s  = *(const float2*)(sinT + t * 32 + ii);
            *(unsigned*)(Ks + row * QSTR + ii) =
                pk2bf(k0.x * c.x - k1.x * s.x, k0.y * c.y - k1.y * s.y);
            *(unsigned*)(Ks + row * QSTR + ii + 32) =
                pk2bf(k1.x * c.x + k0.x * s.x, k1.y * c.y + k0.y * s.y);
        }
        // ---- load V tile transposed: Vt[d][key], packed key-pairs ----
#pragma unroll
        for (int it = 0; it < 8; ++it) {
            int p = tid + it * 256;             // 64 d * 32 key-pairs
            int d = p >> 5, kk = (p & 31) << 1;
            long b0 = (long)(kt * 64 + kk) * QKVD + VOFF + kvh * HD + d;
            float v0 = qkv[b0];
            float v1 = qkv[b0 + QKVD];
            *(unsigned*)(Vt + d * QSTR + kk) = pk2bf(v0, v1);
        }
        __syncthreads();   // Ks/Vt ready (and Qs on first iter)

        // ---- S = (Q * 1/8) K^T : 16 queries x 64 keys per wave ----
        v8f S[4];
#pragma unroll
        for (int nt = 0; nt < 4; ++nt)
#pragma unroll
            for (int r = 0; r < 8; ++r) S[nt][r] = 0.f;
#pragma unroll
        for (int ks = 0; ks < 2; ++ks) {
            v16bf a = frag_a(Qs, QSTR, w * 16, ks * 32);
#pragma unroll
            for (int nt = 0; nt < 4; ++nt) {
                v16bf b = frag_b(Ks, QSTR, nt * 16, ks * 32);
                S[nt] = wmma_bf16(a, b, S[nt]);
            }
        }

        // ---- mask + online softmax (rows live in 16-lane halves) ----
#pragma unroll
        for (int r = 0; r < 8; ++r) {
            const int qi = qb * 128 + w * 16 + r + half * 8;
            float mx = m[r];
#pragma unroll
            for (int nt = 0; nt < 4; ++nt) {
                int kj = kt * 64 + nt * 16 + (lane & 15);
                float sv = S[nt][r];
                bool ok = (kj <= qi) && (qi - kj < SW);
                sv = ok ? sv : -1e30f;
                S[nt][r] = sv;
                mx = fmaxf(mx, sv);
            }
#pragma unroll
            for (int off = 1; off < 16; off <<= 1) mx = fmaxf(mx, __shfl_xor(mx, off, 32));
            const float alpha = __expf(m[r] - mx);
            float rs = 0.f;
#pragma unroll
            for (int nt = 0; nt < 4; ++nt) {
                float pv = __expf(S[nt][r] - mx);
                rs += pv;
                Ps[(w * 16 + r + half * 8) * QSTR + nt * 16 + (lane & 15)] = f2bf(pv);
            }
#pragma unroll
            for (int off = 1; off < 16; off <<= 1) rs += __shfl_xor(rs, off, 32);
            l[r] = l[r] * alpha + rs;
            m[r] = mx;
#pragma unroll
            for (int nt = 0; nt < 4; ++nt) O[nt][r] *= alpha;
        }

        // same-wave LDS RAW (Ps write -> frag read across lanes): wait DS ops
        asm volatile("s_wait_dscnt 0" ::: "memory");

        // ---- O += P V : P is A (16x64 keys), Vt is B (keys x d) ----
#pragma unroll
        for (int ks = 0; ks < 2; ++ks) {
            v16bf a = frag_a(Ps, QSTR, w * 16, ks * 32);
#pragma unroll
            for (int nt = 0; nt < 4; ++nt) {
                v16bf b = frag_b(Vt, QSTR, nt * 16, ks * 32);
                O[nt] = wmma_bf16(a, b, O[nt]);
            }
        }
    }

    // ---- normalize and store attn tile (bf16 for the output GEMM) ----
#pragma unroll
    for (int r = 0; r < 8; ++r) {
        const float inv = 1.f / l[r];
        const int t = qb * 128 + w * 16 + r + half * 8;
#pragma unroll
        for (int nt = 0; nt < 4; ++nt) {
            int d = h * HD + nt * 16 + (lane & 15);
            attn[(long)t * ATTND + d] = f2bf(O[nt][r] * inv);
        }
    }
}

// ---------------- launcher ----------------
extern "C" void kernel_launch(void* const* d_in, const int* in_sizes, int n_in,
                              void* d_out, int out_size, void* d_ws, size_t ws_size,
                              hipStream_t stream) {
    (void)in_sizes; (void)n_in; (void)out_size; (void)ws_size;
    const float* x     = (const float*)d_in[0];
    const float* gam   = (const float*)d_in[1];
    const float* qkv_w = (const float*)d_in[2];
    const float* qkv_b = (const float*)d_in[3];
    const float* out_w = (const float*)d_in[4];
    const float* out_b = (const float*)d_in[5];
    const float* sinks = (const float*)d_in[6];
    float* out = (float*)d_out;

    // workspace layout (16B-aligned offsets)
    char* ws = (char*)d_ws;
    bf16*  tb    = (bf16*)(ws);                       // 1024*2880*2  =  5,898,240
    float* qkv   = (float*)(ws + 5898240);            // 1024*5120*4  = 20,971,520
    float* cosT  = (float*)(ws + 26869760);           // 1024*32*4    =    131,072
    float* sinT  = (float*)(ws + 27000832);           //                  131,072
    bf16*  attn  = (bf16*)(ws + 27131904);            // 1024*4096*2  =  8,388,608
    bf16*  qw_bf = (bf16*)(ws + 35520512);            // 5120*2880*2  = 29,491,200
    bf16*  ow_bf = (bf16*)(ws + 65011712);            // 2880*4096*2  = 23,592,960
                                                      // total ~88.6 MB

    rmsnorm_kernel<<<T_TOK, 256, 0, stream>>>(x, gam, tb);
    rope_tables_kernel<<<(T_TOK * 32 + 255) / 256, 256, 0, stream>>>(cosT, sinT);
    cvt_w_kernel<<<2048, 256, 0, stream>>>(qkv_w, qw_bf, (long)QKVD * HIDDEN / 4);
    cvt_w_kernel<<<2048, 256, 0, stream>>>(out_w, ow_bf, (long)HIDDEN * ATTND / 4);
    gemm_bf16_kernel<<<dim3(QKVD / 128, T_TOK / 128), 256, 0, stream>>>(
        tb, qw_bf, qkv_b, qkv, T_TOK, QKVD, HIDDEN);
    attn_kernel<<<dim3(T_TOK / 128, NH), 256, 0, stream>>>(qkv, cosT, sinT, sinks, attn);
    gemm_bf16_kernel<<<dim3((HIDDEN + 127) / 128, T_TOK / 128), 256, 0, stream>>>(
        attn, ow_bf, out_b, out, T_TOK, HIDDEN, ATTND);
}
